// NgramMultiheadAttention_48352741819030
// MI455X (gfx1250) — compile-verified, matrix operand order checked by reading, not run on hardware
//
#include <hip/hip_runtime.h>
#include <hip/hip_bf16.h>

// ---------------- problem constants ----------------
#define T_SEQ   512
#define BATCHN  4
#define NHEAD   16
#define HDIM    64
#define HID     1024
#define NBUCK   32
#define NGRAMN  2
#define S_TOT   1536          // (1+NGRAM)*T
#define BHN     64            // BATCH*NHEAD
#define MROWS   6144          // S_TOT*BATCH

typedef __attribute__((ext_vector_type(16))) __bf16 v16bf;
typedef __attribute__((ext_vector_type(8)))  float  v8f;

static __device__ __forceinline__ v8f wmma_bf16(v16bf a, v16bf b, v8f c) {
  // D = A(16x32 bf16) * B(32x16 bf16) + C(16x16 f32)
  return __builtin_amdgcn_wmma_f32_16x16x32_bf16(false, a, false, b, (short)0, c,
                                                 false, false);
}

// Per-lane 16-bit A-operand K layout (ISA 7.12.2):
//   lanes 0-15 : elems 0..7 -> K = 0..7,  elems 8..15 -> K = 16..23
//   lanes 16-31: elems 0..7 -> K = 8..15, elems 8..15 -> K = 24..31
// i.e. chunk0 at K = hi*8, chunk1 at K = 16 + hi*8 (8 contiguous each).

// ============================================================================
// Generic WMMA GEMM: C[m,n] = (sum_k A_f32[m,k] * W_f32[n,k] + bias[n]) * scale
// A: MROWS x K row-major f32, W: N x K row-major f32.
// One wave computes a 16(M) x 64(N) tile (4 WMMA accumulators, A reused 4x).
// MODE: 0 = q/k store bf16 [bh][s][d]; 2 = v store bf16 transposed [bh][d][s];
//       3 = rel scatter f32;            4 = f32 row-major to d_out.
// ============================================================================
template<int MODE>
__global__ void gemm_wmma(const float* __restrict__ A,
                          const float* __restrict__ W,
                          const float* __restrict__ bias,
                          int N, int K, float scale,
                          __bf16* __restrict__ dst_bf,
                          float*  __restrict__ dst_f0,
                          float*  __restrict__ dst_f1) {
  const int wave = threadIdx.x >> 5;
  const int lane = threadIdx.x & 31;
  const int r    = lane & 15;
  const int hi   = lane >> 4;
  const int ngroups = N >> 6;
  const int w     = blockIdx.x * (blockDim.x >> 5) + wave;
  const int mtile = w / ngroups;
  const int ng    = w % ngroups;
  if (mtile * 16 >= MROWS) return;

  const float* arow = A + (size_t)(mtile * 16 + r) * K;
  v8f acc[4] = {v8f{}, v8f{}, v8f{}, v8f{}};

  for (int k0 = 0; k0 < K; k0 += 32) {
    v16bf a;
    const float* pa = arow + k0;
#pragma unroll
    for (int j = 0; j < 8; ++j) {
      a[j]     = (__bf16)pa[hi * 8 + j];
      a[8 + j] = (__bf16)pa[16 + hi * 8 + j];
    }
#pragma unroll
    for (int jt = 0; jt < 4; ++jt) {
      const int n = ng * 64 + jt * 16 + r;
      const float* pw = W + (size_t)n * K + k0;
      v16bf bm;
#pragma unroll
      for (int j = 0; j < 8; ++j) {
        bm[j]     = (__bf16)pw[hi * 8 + j];
        bm[8 + j] = (__bf16)pw[16 + hi * 8 + j];
      }
      acc[jt] = wmma_bf16(a, bm, acc[jt]);
    }
  }

#pragma unroll
  for (int jt = 0; jt < 4; ++jt) {
    const int n  = ng * 64 + jt * 16 + r;
    const float bv = bias[n];
#pragma unroll
    for (int i = 0; i < 8; ++i) {
      const int m = mtile * 16 + i + hi * 8;
      const float val = (acc[jt][i] + bv) * scale;
      const int s = m >> 2;        // hidden is (S, B, HID): row m = s*B + b
      const int b = m & 3;
      if (MODE == 0) {                               // q / k : [bh][s][d] bf16
        const int h = n >> 6, d = n & 63;
        dst_bf[(((size_t)(b * NHEAD + h)) * S_TOT + s) * HDIM + d] = (__bf16)val;
      } else if (MODE == 2) {                        // v transposed [bh][d][s]
        const int h = n >> 6, d = n & 63;
        dst_bf[(((size_t)(b * NHEAD + h)) * HDIM + d) * S_TOT + s] = (__bf16)val;
      } else if (MODE == 3) {                        // rel: feature n = nb*H + h
        const int h = n & 15, nb = n >> 4;
        if (s < T_SEQ) {
          dst_f0[(((size_t)(b * NHEAD + h)) * T_SEQ + s) * NBUCK + nb] = val;
        } else {
          const int g = s / T_SEQ - 1, t = s % T_SEQ;
          dst_f1[((((size_t)(g * BATCHN + b)) * NHEAD + h) * T_SEQ + t) * NBUCK + nb] = val;
        }
      } else {                                       // out-projection f32
        dst_f0[(size_t)m * HID + n] = val;
      }
    }
  }
}

// ============================================================================
// Main-stream attention: per wave 16 query rows, full 512-key softmax in LDS.
// block = 64 threads (2 waves), dynamic LDS = 2*16*512 f32 = 64 KB.
// grid = BH * (T/32) = 1024.
// ============================================================================
__global__ void attn_main_kernel(const __bf16* __restrict__ qb,
                                 const __bf16* __restrict__ kb,
                                 const __bf16* __restrict__ vT,
                                 const float* __restrict__ relm,
                                 const int*   __restrict__ buckets,
                                 const float* __restrict__ mask,
                                 float* __restrict__ probs_out,
                                 float* __restrict__ ctx) {
  extern __shared__ float smem[];
  const int wave = threadIdx.x >> 5;
  const int lane = threadIdx.x & 31;
  const int r = lane & 15, hi = lane >> 4;
  const int bh   = blockIdx.x >> 4;
  const int tblk = blockIdx.x & 15;
  const int t0   = tblk * 32 + wave * 16;
  const int b = bh >> 4, h = bh & 15;
  float* sc = smem + (size_t)wave * 16 * 512;

  // Q operands for rows t0..t0+15, K (=d) 0..63 -> two 16x32 A tiles
  const __bf16* qrow = qb + ((size_t)bh * S_TOT + t0 + r) * HDIM;
  v16bf a0, a1;
#pragma unroll
  for (int j = 0; j < 8; ++j) {
    a0[j]     = qrow[hi * 8 + j];
    a0[8 + j] = qrow[16 + hi * 8 + j];
    a1[j]     = qrow[32 + hi * 8 + j];
    a1[8 + j] = qrow[48 + hi * 8 + j];
  }
  // scores = Q K^T into LDS
  for (int st = 0; st < 32; ++st) {
    const int s0 = st * 16;
    const __bf16* krow = kb + ((size_t)bh * S_TOT + s0 + r) * HDIM;
    v16bf b0, b1;
#pragma unroll
    for (int j = 0; j < 8; ++j) {
      b0[j]     = krow[hi * 8 + j];
      b0[8 + j] = krow[16 + hi * 8 + j];
      b1[j]     = krow[32 + hi * 8 + j];
      b1[8 + j] = krow[48 + hi * 8 + j];
    }
    v8f c = {};
    c = wmma_bf16(a0, b0, c);
    c = wmma_bf16(a1, b1, c);
#pragma unroll
    for (int i = 0; i < 8; ++i)
      sc[(i + hi * 8) * 512 + s0 + r] = c[i];
  }
  __syncthreads();

  // softmax: lane pair (r, r+16) owns row r; each half scans 256 keys
  {
    const int t = t0 + r;
    const int*   brow   = buckets + ((size_t)b * T_SEQ + t) * T_SEQ;
    const float* relrow = relm + ((size_t)bh * T_SEQ + t) * NBUCK;
    const float* mrow   = mask + ((size_t)bh * T_SEQ + t) * T_SEQ;
    float* srow = sc + r * 512;
    const int sbeg = hi * 256, send = sbeg + 256;
    float mx = -3.0e38f;
    for (int s = sbeg; s < send; ++s) {
      const float v = srow[s] + relrow[brow[s]] + mrow[s];
      srow[s] = v;
      mx = fmaxf(mx, v);
    }
    mx = fmaxf(mx, __shfl_xor(mx, 16, 32));
    float sum = 0.f;
    for (int s = sbeg; s < send; ++s) {
      const float e = __expf(srow[s] - mx);
      srow[s] = e;
      sum += e;
    }
    sum += __shfl_xor(sum, 16, 32);
    const float inv = 1.0f / sum;
    float* pout = probs_out + ((size_t)bh * T_SEQ + t) * T_SEQ;
    for (int s = sbeg; s < send; ++s) {
      const float p = srow[s] * inv;
      srow[s] = p;
      pout[s] = p;
    }
  }
  __syncthreads();

  // PV: out(16x64) = probs(16x512) @ V(512x64), V stored transposed [bh][d][s]
  v8f o[4] = {v8f{}, v8f{}, v8f{}, v8f{}};
  for (int kt = 0; kt < 16; ++kt) {
    const int kbase = kt * 32;
    v16bf a;
#pragma unroll
    for (int j = 0; j < 8; ++j) {
      a[j]     = (__bf16)sc[r * 512 + kbase + hi * 8 + j];
      a[8 + j] = (__bf16)sc[r * 512 + kbase + 16 + hi * 8 + j];
    }
#pragma unroll
    for (int jt = 0; jt < 4; ++jt) {
      const int d = jt * 16 + r;
      const __bf16* vrow = vT + ((size_t)bh * HDIM + d) * S_TOT + kbase;
      v16bf bm;
#pragma unroll
      for (int j = 0; j < 8; ++j) {
        bm[j]     = vrow[hi * 8 + j];
        bm[8 + j] = vrow[16 + hi * 8 + j];
      }
      o[jt] = wmma_bf16(a, bm, o[jt]);
    }
  }
#pragma unroll
  for (int jt = 0; jt < 4; ++jt) {
#pragma unroll
    for (int i = 0; i < 8; ++i) {
      const int t = t0 + i + hi * 8;
      const int d = jt * 16 + r;
      ctx[((size_t)t * BATCHN + b) * HID + h * HDIM + d] = o[jt][i];
    }
  }
}

// ============================================================================
// Predict-stream attention: keys = concat(main, gram segment) -> 1024 keys.
// block = 64 threads (2 waves), dynamic LDS = 2*16*1024 f32 = 128 KB.
// grid = NGRAM * BH * (T/32) = 2048.
// ============================================================================
__global__ void attn_pred_kernel(const __bf16* __restrict__ qb,
                                 const __bf16* __restrict__ kb,
                                 const __bf16* __restrict__ vT,
                                 const float* __restrict__ relp,
                                 const int*   __restrict__ bucketsP,
                                 const float* __restrict__ emask,
                                 float* __restrict__ probsP,
                                 float* __restrict__ ctx) {
  extern __shared__ float smem[];
  const int wave = threadIdx.x >> 5;
  const int lane = threadIdx.x & 31;
  const int r = lane & 15, hi = lane >> 4;
  const int ng  = blockIdx.x >> 10;
  const int rem = blockIdx.x & 1023;
  const int bh   = rem >> 4;
  const int tblk = rem & 15;
  const int t0   = tblk * 32 + wave * 16;
  const int b = bh >> 4, h = bh & 15;
  float* sc = smem + (size_t)wave * 16 * 1024;

  const __bf16* qrow =
      qb + ((size_t)bh * S_TOT + (ng + 1) * T_SEQ + t0 + r) * HDIM;
  v16bf a0, a1;
#pragma unroll
  for (int j = 0; j < 8; ++j) {
    a0[j]     = qrow[hi * 8 + j];
    a0[8 + j] = qrow[16 + hi * 8 + j];
    a1[j]     = qrow[32 + hi * 8 + j];
    a1[8 + j] = qrow[48 + hi * 8 + j];
  }
  for (int st = 0; st < 64; ++st) {
    const int s0 = st * 16;
    const int skey = s0 + r;
    const int seq = (skey < T_SEQ) ? skey : (ng + 1) * T_SEQ + (skey - T_SEQ);
    const __bf16* krow = kb + ((size_t)bh * S_TOT + seq) * HDIM;
    v16bf b0, b1;
#pragma unroll
    for (int j = 0; j < 8; ++j) {
      b0[j]     = krow[hi * 8 + j];
      b0[8 + j] = krow[16 + hi * 8 + j];
      b1[j]     = krow[32 + hi * 8 + j];
      b1[8 + j] = krow[48 + hi * 8 + j];
    }
    v8f c = {};
    c = wmma_bf16(a0, b0, c);
    c = wmma_bf16(a1, b1, c);
#pragma unroll
    for (int i = 0; i < 8; ++i)
      sc[(i + hi * 8) * 1024 + s0 + r] = c[i];
  }
  __syncthreads();

  {
    const int t = t0 + r;
    const int*   brow   = bucketsP + ((size_t)b * T_SEQ + t) * 1024;
    const float* relrow =
        relp + ((((size_t)ng * BATCHN + b) * NHEAD + h) * T_SEQ + t) * NBUCK;
    const float* mrow = emask + (((size_t)ng * BHN + bh) * T_SEQ + t) * 1024;
    float* srow = sc + r * 1024;
    const int sbeg = hi * 512, send = sbeg + 512;
    float mx = -3.0e38f;
    for (int s = sbeg; s < send; ++s) {
      const float v = srow[s] + relrow[brow[s]] + mrow[s];
      srow[s] = v;
      mx = fmaxf(mx, v);
    }
    mx = fmaxf(mx, __shfl_xor(mx, 16, 32));
    float sum = 0.f;
    for (int s = sbeg; s < send; ++s) {
      const float e = __expf(srow[s] - mx);
      srow[s] = e;
      sum += e;
    }
    sum += __shfl_xor(sum, 16, 32);
    const float inv = 1.0f / sum;
    // output layout (B, NGRAM, H, T, 2T)
    float* pout =
        probsP + ((((size_t)b * NGRAMN + ng) * NHEAD + h) * T_SEQ + t) * 1024;
    for (int s = sbeg; s < send; ++s) {
      const float p = srow[s] * inv;
      srow[s] = p;
      pout[s] = p;
    }
  }
  __syncthreads();

  v8f o[4] = {v8f{}, v8f{}, v8f{}, v8f{}};
  for (int kt = 0; kt < 32; ++kt) {
    const int kbase = kt * 32;
    v16bf a;
#pragma unroll
    for (int j = 0; j < 8; ++j) {
      a[j]     = (__bf16)sc[r * 1024 + kbase + hi * 8 + j];
      a[8 + j] = (__bf16)sc[r * 1024 + kbase + 16 + hi * 8 + j];
    }
#pragma unroll
    for (int jt = 0; jt < 4; ++jt) {
      const int d = jt * 16 + r;
      const __bf16* vbase = vT + ((size_t)bh * HDIM + d) * S_TOT;
      v16bf bm;
#pragma unroll
      for (int j = 0; j < 8; ++j) {
        int s1 = kbase + hi * 8 + j;
        int s2 = kbase + 16 + hi * 8 + j;
        s1 = (s1 < T_SEQ) ? s1 : (ng + 1) * T_SEQ + (s1 - T_SEQ);
        s2 = (s2 < T_SEQ) ? s2 : (ng + 1) * T_SEQ + (s2 - T_SEQ);
        bm[j]     = vbase[s1];
        bm[8 + j] = vbase[s2];
      }
      o[jt] = wmma_bf16(a, bm, o[jt]);
    }
  }
#pragma unroll
  for (int jt = 0; jt < 4; ++jt) {
#pragma unroll
    for (int i = 0; i < 8; ++i) {
      const int t = t0 + i + hi * 8;
      const int d = jt * 16 + r;
      const int sg = (ng + 1) * T_SEQ + t;
      ctx[((size_t)sg * BATCHN + b) * HID + h * HDIM + d] = o[jt][i];
    }
  }
}

// ============================================================================
extern "C" void kernel_launch(void* const* d_in, const int* in_sizes, int n_in,
                              void* d_out, int out_size, void* d_ws,
                              size_t ws_size, hipStream_t stream) {
  const float* hidden = (const float*)d_in[0];
  const float* amask  = (const float*)d_in[1];
  const float* emask  = (const float*)d_in[2];
  const int*   mbuck  = (const int*)d_in[3];
  const int*   pbuck  = (const int*)d_in[4];
  const float* Wq = (const float*)d_in[5];   const float* bq = (const float*)d_in[6];
  const float* Wk = (const float*)d_in[7];   const float* bk = (const float*)d_in[8];
  const float* Wv = (const float*)d_in[9];   const float* bv = (const float*)d_in[10];
  const float* Wo = (const float*)d_in[11];  const float* bo = (const float*)d_in[12];
  const float* Wr = (const float*)d_in[13];  const float* br = (const float*)d_in[14];

  char* ws = (char*)d_ws;
  __bf16* qb   = (__bf16*)ws;  ws += (size_t)BHN * S_TOT * HDIM * 2;
  __bf16* kbuf = (__bf16*)ws;  ws += (size_t)BHN * S_TOT * HDIM * 2;
  __bf16* vT   = (__bf16*)ws;  ws += (size_t)BHN * S_TOT * HDIM * 2;
  float*  relm = (float*)ws;   ws += (size_t)BHN * T_SEQ * NBUCK * 4;
  float*  relp = (float*)ws;   ws += (size_t)NGRAMN * BHN * T_SEQ * NBUCK * 4;
  float*  ctx  = (float*)ws;   ws += (size_t)MROWS * HID * 4;

  float* out_attn = (float*)d_out;
  float* out_pm   = out_attn + (size_t)S_TOT * BATCHN * HID;
  float* out_pp   = out_pm + (size_t)BATCHN * NHEAD * T_SEQ * T_SEQ;

  // projections (bf16 WMMA, f32 accumulate)
  gemm_wmma<0><<<1536, 128, 0, stream>>>(hidden, Wq, bq, 1024, 1024, 0.125f,
                                         qb, nullptr, nullptr);
  gemm_wmma<0><<<1536, 128, 0, stream>>>(hidden, Wk, bk, 1024, 1024, 1.0f,
                                         kbuf, nullptr, nullptr);
  gemm_wmma<2><<<1536, 128, 0, stream>>>(hidden, Wv, bv, 1024, 1024, 1.0f,
                                         vT, nullptr, nullptr);
  gemm_wmma<3><<<768, 128, 0, stream>>>(hidden, Wr, br, 512, 1024, 1.0f,
                                        nullptr, relm, relp);
  // fused attention (scores + bucket gather + softmax + PV)
  attn_main_kernel<<<1024, 64, 64 * 1024, stream>>>(qb, kbuf, vT, relm, mbuck,
                                                    amask, out_pm, ctx);
  attn_pred_kernel<<<2048, 64, 128 * 1024, stream>>>(qb, kbuf, vT, relp, pbuck,
                                                     emask, out_pp, ctx);
  // output projection
  gemm_wmma<4><<<1536, 128, 0, stream>>>(ctx, Wo, bo, 1024, 1024, 1.0f,
                                         nullptr, out_attn, nullptr);
}
